// NewsClassifierWithRNN_78451872629442
// MI455X (gfx1250) — matrix operand order repaired and checked by reading, not compiled, not run on hardware
//
#include <hip/hip_runtime.h>

// Problem constants (from reference)
constexpr int B = 128, S = 512, V = 100000, E = 128, H = 256, C = 4;

typedef __attribute__((ext_vector_type(16))) __bf16 v16bf;
typedef __attribute__((ext_vector_type(8)))  float  v8f;

union Frag32B { uint4 u[2]; v16bf v; };

__device__ inline unsigned short f2bf(float f) {
    unsigned int u = __float_as_uint(f);
    u += 0x7FFFu + ((u >> 16) & 1u);          // round-to-nearest-even
    return (unsigned short)(u >> 16);
}

// ---------------------------------------------------------------------------
// Kernel 1: convert w_ih [H,E] and w_hh [H,H] f32 -> bf16 in workspace
// ---------------------------------------------------------------------------
__global__ void k_convert(const float* __restrict__ w_ih, const float* __restrict__ w_hh,
                          unsigned short* __restrict__ wih_bf, unsigned short* __restrict__ whh_bf) {
    int i = blockIdx.x * 256 + threadIdx.x;
    if (i < H * E) wih_bf[i] = f2bf(w_ih[i]);
    int j = i - H * E;
    if (j >= 0 && j < H * H) whh_bf[j] = f2bf(w_hh[j]);
}

// ---------------------------------------------------------------------------
// Kernel 2: embedding gather + input projection (memory-bound, ~100 MB total).
//   Z[s][b][n] = sum_e table[x[b][s]][e]*w_ih[n][e] + b_ih[n] + b_hh[n]
// One block = 32 rows (fixed b, 32 consecutive s) x all 256 columns.
// 8 waves: 2 M-tiles x 4 N-groups(64 cols each); K = 128 = 4 WMMA steps.
// ---------------------------------------------------------------------------
__global__ __launch_bounds__(256) void k_embed_proj(
        const long long* __restrict__ x, const float* __restrict__ emb,
        const unsigned short* __restrict__ wih_bf,
        const float* __restrict__ b_ih, const float* __restrict__ b_hh,
        float* __restrict__ Z) {
    extern __shared__ unsigned char smem[];
    unsigned short* sWih = (unsigned short*)smem;            // [256][128] bf16 (64 KB)
    unsigned short* sA   = (unsigned short*)(smem + 65536);  // [32][128]  bf16 (8 KB)

    const int m0    = blockIdx.x * 32;
    const int b_idx = m0 / S;
    const int s0    = m0 % S;        // 32 | S, so b is constant in the block

    // stage W_ih (bf16) into LDS
    { const uint4* src = (const uint4*)wih_bf; uint4* dst = (uint4*)sWih;
      for (int i = threadIdx.x; i < (H * E) / 8; i += 256) dst[i] = src[i]; }

    // gather+convert 32 embedding rows (8 threads/row, 16 floats/thread)
    { int r = threadIdx.x >> 3, p = threadIdx.x & 7;
      long long tok = x[(size_t)b_idx * S + s0 + r];
      unsigned int* dst = (unsigned int*)&sA[r * 128 + p * 16];
      if (tok == 0) {                               // padding_idx = 0 -> zero row
          for (int q = 0; q < 8; ++q) dst[q] = 0u;
      } else {
          const float4* rp = (const float4*)(emb + (size_t)tok * E + p * 16);
          for (int q = 0; q < 4; ++q) {
              float4 f = rp[q];
              dst[q * 2 + 0] = (unsigned int)f2bf(f.x) | ((unsigned int)f2bf(f.y) << 16);
              dst[q * 2 + 1] = (unsigned int)f2bf(f.z) | ((unsigned int)f2bf(f.w) << 16);
          }
      }
    }
    __syncthreads();

    const int wv = threadIdx.x >> 5, lane = threadIdx.x & 31;
    const int mt = wv >> 2, ng = wv & 3;
    const int lr = lane & 15;
    const bool hiL = lane >= 16;
    const int arow = mt * 16 + lr;

    v8f acc[4] = {};
    for (int kk = 0; kk < 4; ++kk) {
        const int k0 = kk * 32;
        Frag32B a;                                   // A: 16x32 bf16 (ISA 7.12.2 layout)
        const int kb = k0 + (hiL ? 8 : 0);
        a.u[0] = *(const uint4*)&sA[arow * 128 + kb];
        a.u[1] = *(const uint4*)&sA[arow * 128 + kb + 16];
        const int kb2 = k0 + (hiL ? 16 : 0);
        for (int t = 0; t < 4; ++t) {
            const int ncol = ng * 64 + t * 16 + lr;  // B[k][n] = w_ih[n][k] (row-contig)
            Frag32B bf;
            const uint4* bp = (const uint4*)&sWih[ncol * 128 + kb];
            bf.u[0] = bp[0]; bf.u[1] = bp[2];
            acc[t] = __builtin_amdgcn_wmma_f32_16x16x32_bf16(
                false, a.v, false, bf.v, (short)0, acc[t], false, false);
        }
    }
    // bias add + store Z[s][b][n]
    for (int t = 0; t < 4; ++t) {
        const int ncol = ng * 64 + t * 16 + lr;
        const float bias = b_ih[ncol] + b_hh[ncol];
        for (int r = 0; r < 8; ++r) {
            const int mloc = mt * 16 + (hiL ? 8 : 0) + r;
            const int s = s0 + mloc;
            Z[((size_t)s * B + b_idx) * H + ncol] = acc[t][r] + bias;
        }
    }
}

// ---------------------------------------------------------------------------
// Kernel 3: sequential RNN scan, batch-split across 8 WGPs.
//   h_t[b] = tanh(Z[t][b] + h_{t-1}[b] @ W_hh^T)   -- rows are independent!
// Each block owns 16 batch rows (= one WMMA M-tile) for all 512 steps.
// W_hh is REGISTER-RESIDENT: each wave owns one 16-col n-tile, whose 8
// K-fragments (64 VGPRs) are loaded from global once and reused 512 times.
// LDS/block: double-buffered h only (2 x 8 KB). Inner loop per step per wave:
// 16 A-loads (ds_load_b128) + 8 chained WMMAs + 8 Z loads + tanh + 8 b16 stores.
// ---------------------------------------------------------------------------
__global__ __launch_bounds__(512) void k_scan(
        const unsigned short* __restrict__ whh_bf, const float* __restrict__ Z,
        float* __restrict__ hlast) {
    extern __shared__ unsigned char smem[];
    unsigned short* sH = (unsigned short*)smem;              // [2][16][256] bf16

    { uint4 z = {0u, 0u, 0u, 0u}; uint4* dst = (uint4*)sH;   // h0 = 0 (buffer 0 only, read at step 0)
      for (int i = threadIdx.x; i < (16 * H) / 8; i += 512) dst[i] = z; }

    const int wv = threadIdx.x >> 5, lane = threadIdx.x & 31;  // wv = n-tile 0..15
    const int lr = lane & 15;
    const bool hiL = lane >= 16;
    const int ncol  = wv * 16 + lr;
    const int brow0 = blockIdx.x * 16;                         // this WGP's batch rows

    // hoist loop-invariant W_hh fragments into registers (B[k][n] = w_hh[n][k])
    Frag32B bfr[8];
#pragma unroll
    for (int kk = 0; kk < 8; ++kk) {
        const uint4* bp = (const uint4*)&whh_bf[ncol * 256 + kk * 32 + (hiL ? 8 : 0)];
        bfr[kk].u[0] = bp[0];
        bfr[kk].u[1] = bp[2];
    }
    __syncthreads();

    for (int s = 0; s < S; ++s) {
        const unsigned short* hc = sH + (s & 1) * (16 * H);
        unsigned short* hn = sH + ((s & 1) ^ 1) * (16 * H);
        const float* Zs = Z + (size_t)s * (B * H) + (size_t)brow0 * H;  // 16 KB slab
        if (s + 1 < S)   // pull next step's slab toward L2/WGP$ while computing
            __builtin_prefetch(Zs + (size_t)B * H + threadIdx.x * 8, 0, 1);

        v8f acc = {};
#pragma unroll
        for (int kk = 0; kk < 8; ++kk) {
            Frag32B a;                                // A = h tile (ISA 7.12.2 layout)
            const int kb = kk * 32 + (hiL ? 8 : 0);
            a.u[0] = *(const uint4*)&hc[lr * 256 + kb];
            a.u[1] = *(const uint4*)&hc[lr * 256 + kb + 16];
            acc = __builtin_amdgcn_wmma_f32_16x16x32_bf16(
                false, a.v, false, bfr[kk].v, (short)0, acc, false, false);
        }
#pragma unroll
        for (int r = 0; r < 8; ++r) {
            const int m = (hiL ? 8 : 0) + r;
            const float v = acc[r] + Zs[(size_t)m * H + ncol];
            const float h = tanhf(v);
            hn[m * 256 + ncol] = f2bf(h);
            if (s == S - 1) hlast[(size_t)(brow0 + m) * H + ncol] = h;
        }
        __syncthreads();   // separates step s (read cur, write nxt) from step s+1
    }
}

// ---------------------------------------------------------------------------
// Kernel 4: classifier head in full f32 (tiny: 17 MFLOP).
//   out = relu(h @ w1^T + b1) @ w2^T + b2
// ---------------------------------------------------------------------------
__global__ __launch_bounds__(1024) void k_head(
        const float* __restrict__ hlast, const float* __restrict__ w1,
        const float* __restrict__ b1, const float* __restrict__ w2,
        const float* __restrict__ b2, float* __restrict__ out) {
    extern __shared__ unsigned char smem[];
    float* sHL = (float*)smem;             // [k][b] transposed, conflict-free reads
    float* sG  = (float*)(smem + 131072);  // [j][b]
    for (int i = threadIdx.x; i < B * H; i += 1024) {
        int b = i >> 8, k = i & 255;
        sHL[k * 128 + b] = hlast[i];
    }
    __syncthreads();
    for (int i = threadIdx.x; i < B * H; i += 1024) {
        int b = i & 127, j = i >> 7;       // whole wave shares j -> w1 broadcast
        float acc = b1[j];
        const float* wr = w1 + j * 256;
        for (int k = 0; k < 256; ++k) acc = fmaf(sHL[k * 128 + b], wr[k], acc);
        sG[j * 128 + b] = fmaxf(acc, 0.0f);
    }
    __syncthreads();
    if (threadIdx.x < B * C) {
        int b = threadIdx.x >> 2, c = threadIdx.x & 3;
        float acc = b2[c];
        const float* wr = w2 + c * 256;
        for (int j = 0; j < 256; ++j) acc = fmaf(sG[j * 128 + b], wr[j], acc);
        out[b * C + c] = acc;
    }
}

// ---------------------------------------------------------------------------
extern "C" void kernel_launch(void* const* d_in, const int* in_sizes, int n_in,
                              void* d_out, int out_size, void* d_ws, size_t ws_size,
                              hipStream_t stream) {
    (void)in_sizes; (void)n_in; (void)out_size; (void)ws_size;
    const long long* x    = (const long long*)d_in[0];
    const float* emb      = (const float*)d_in[1];
    const float* w_ih     = (const float*)d_in[2];
    const float* b_ih     = (const float*)d_in[3];
    const float* w_hh     = (const float*)d_in[4];
    const float* b_hh     = (const float*)d_in[5];
    const float* w1       = (const float*)d_in[6];
    const float* b1       = (const float*)d_in[7];
    const float* w2       = (const float*)d_in[8];
    const float* b2       = (const float*)d_in[9];
    float* out = (float*)d_out;

    // workspace layout
    char* ws = (char*)d_ws;
    unsigned short* wih_bf = (unsigned short*)(ws);             // 64 KB
    unsigned short* whh_bf = (unsigned short*)(ws + 65536);     // 128 KB
    float* hlast           = (float*)(ws + 196608);             // 128 KB
    float* Z               = (float*)(ws + 327680);             // 64 MB [S][B][H]

    k_convert   <<<384, 256, 0, stream>>>(w_ih, w_hh, wih_bf, whh_bf);
    k_embed_proj<<<(B * S) / 32, 256, 73728, stream>>>(x, emb, wih_bf, b_ih, b_hh, Z);
    k_scan      <<<B / 16, 512, 16384, stream>>>(whh_bf, Z, hlast);
    k_head      <<<1, 1024, 262144, stream>>>(hlast, w1, b1, w2, b2, out);
}
